// MoELayer_25769803776018
// MI455X (gfx1250) — matrix operand order repaired and checked by reading, not compiled, not run on hardware
//
#include <hip/hip_runtime.h>
#include <cstdint>
#include <cstddef>

// ---------------------------------------------------------------------------
// MoE forward for MI455X (gfx1250): bf16 WMMA grouped GEMM, wave32.
//   block tile 128(M) x 256(N), K-step 32; 8 waves as 2(M) x 4(N);
//   each wave: 4x4 fragments -> 16 v_wmma_f32_16x16x32_bf16 per K-step.
// ---------------------------------------------------------------------------

#define H_DIM 2048
#define E_NUM 8
#define T_NUM 8192   // B*S = 4*2048

typedef __attribute__((ext_vector_type(16))) __bf16    v16bf;
typedef __attribute__((ext_vector_type(2)))  __bf16    bf2v;
typedef __attribute__((ext_vector_type(8)))  float     v8f;
typedef __attribute__((ext_vector_type(4)))  float     f4v;
typedef __attribute__((ext_vector_type(2)))  float     f2v;
typedef __attribute__((ext_vector_type(4)))  unsigned  u4v;

union FragAB { v16bf v; u4v q[2]; };

// f32x2 -> packed bf16x2 (RTNE) -- should lower to v_cvt_pk_bf16_f32
__device__ __forceinline__ unsigned pack_bf16(float a, float b) {
  f2v f; f.x = a; f.y = b;
  bf2v h = __builtin_convertvector(f, bf2v);
  union { bf2v v; unsigned u; } c; c.v = h;
  return c.u;
}

// ---------------------------------------------------------------------------
// 1) zero output accumulator + expert counters
// ---------------------------------------------------------------------------
__global__ __launch_bounds__(256) void moe_zero_kernel(float* __restrict__ out,
                                                       int* __restrict__ counts,
                                                       long long n4) {
  long long i = (long long)blockIdx.x * blockDim.x + threadIdx.x;
  if (blockIdx.x == 0 && threadIdx.x < E_NUM) counts[threadIdx.x] = 0;
  if (i < n4) {
    f4v z = {};
    ((f4v*)out)[i] = z;
  }
}

// ---------------------------------------------------------------------------
// 2) router: one wave32 per token; 8 logits, exact top-2 softmax renorm
// ---------------------------------------------------------------------------
__global__ __launch_bounds__(256) void moe_router_kernel(
    const float* __restrict__ x, const float* __restrict__ rw,
    const float* __restrict__ rb, int* __restrict__ assign_e,
    float* __restrict__ assign_g, int* __restrict__ counts) {
  const int wave = threadIdx.x >> 5;
  const int lane = threadIdx.x & 31;
  const int t = blockIdx.x * 8 + wave;
  const float* xp = x + (size_t)t * H_DIM;

  float acc[E_NUM];
#pragma unroll
  for (int e = 0; e < E_NUM; ++e) acc[e] = 0.f;

  for (int h = lane; h < H_DIM; h += 32) {
    float xv = xp[h];
#pragma unroll
    for (int e = 0; e < E_NUM; ++e) acc[e] += xv * rw[e * H_DIM + h];
  }

  float lg[E_NUM];
#pragma unroll
  for (int e = 0; e < E_NUM; ++e) {
    float v = acc[e];
#pragma unroll
    for (int off = 16; off > 0; off >>= 1) v += __shfl_xor(v, off, 32);
    lg[e] = v + rb[e];
  }

  if (lane == 0) {
    int e0 = 0; float m0 = lg[0];
#pragma unroll
    for (int e = 1; e < E_NUM; ++e) if (lg[e] > m0) { m0 = lg[e]; e0 = e; }
    int e1 = (e0 == 0) ? 1 : 0; float m1 = lg[e1];
#pragma unroll
    for (int e = 0; e < E_NUM; ++e)
      if (e != e0 && lg[e] > m1) { m1 = lg[e]; e1 = e; }
    // softmax over {m0,m1} then renormalize -> gates sum to exactly 1
    float z = __expf(m1 - m0);
    float inv = 1.f / (1.f + z);
    assign_e[2 * t]     = e0;
    assign_e[2 * t + 1] = e1;
    assign_g[2 * t]     = inv;
    assign_g[2 * t + 1] = z * inv;
    atomicAdd(&counts[e0], 1);
    atomicAdd(&counts[e1], 1);
  }
}

// ---------------------------------------------------------------------------
// 3) exclusive scan over E=8 counts
// ---------------------------------------------------------------------------
__global__ void moe_scan_kernel(const int* __restrict__ counts,
                                int* __restrict__ offsets,
                                int* __restrict__ cursor) {
  if (threadIdx.x == 0) {
    int s = 0;
    for (int e = 0; e < E_NUM; ++e) { offsets[e] = s; cursor[e] = s; s += counts[e]; }
  }
}

// ---------------------------------------------------------------------------
// 4) scatter tokens into per-expert buckets
// ---------------------------------------------------------------------------
__global__ __launch_bounds__(256) void moe_scatter_kernel(
    const int* __restrict__ assign_e, const float* __restrict__ assign_g,
    int* __restrict__ cursor, int* __restrict__ btok, float* __restrict__ bgate) {
  int t = blockIdx.x * blockDim.x + threadIdx.x;
  if (t >= T_NUM) return;
#pragma unroll
  for (int k = 0; k < 2; ++k) {
    int e = assign_e[2 * t + k];
    int slot = atomicAdd(&cursor[e], 1);
    btok[slot]  = t;
    bgate[slot] = assign_g[2 * t + k];
  }
}

// ---------------------------------------------------------------------------
// 5) grouped GEMM: per expert, Y[rows, n0:n0+256] += gate * (X_rows @ W_e^T)
// ---------------------------------------------------------------------------
__global__ __launch_bounds__(256) void moe_gemm_kernel(
    const float* __restrict__ x, const float* __restrict__ ew,
    const int* __restrict__ counts, const int* __restrict__ offsets,
    const int* __restrict__ btok, const float* __restrict__ bgate,
    float* __restrict__ out) {
  const int e  = blockIdx.z;
  const int ne = counts[e];
  const int m0 = blockIdx.y * 128;
  if (m0 >= ne) return;                       // block-uniform early exit
  const int n0    = blockIdx.x * 256;
  const int off   = offsets[e];
  const int mrows = min(128, ne - m0);

  // LDS tiles: [row][K=32] bf16, stride padded to 40 elems (80B, 16B-aligned)
  __shared__ unsigned ldsA[128 * 20];
  __shared__ unsigned ldsB[256 * 20];
  __shared__ int   s_tok[128];
  __shared__ float s_gate[128];

  const int tid = threadIdx.x;
  if (tid < 128) {
    if (tid < mrows) {
      s_tok[tid]  = btok[off + m0 + tid];
      s_gate[tid] = bgate[off + m0 + tid];
    } else {
      s_tok[tid] = 0; s_gate[tid] = 0.f;
    }
  }
  __syncthreads();

  // A staging: 2 threads per token row, 16 f32 each.  B staging: 1 thread per
  // weight row (n0+tid), 32 f32 each (contiguous: row of W_e).
  const int arow = tid >> 1;                  // 0..127
  const int kh   = (tid & 1) * 16;            // 0 or 16 (elements)
  const float* xrow = x + (size_t)s_tok[arow] * H_DIM + kh;
  const float* wrow = ew + (size_t)e * H_DIM * H_DIM + (size_t)(n0 + tid) * H_DIM;
  const bool arow_valid = arow < mrows;

  const int wave = tid >> 5;
  const int lane = tid & 31;
  const int half = lane >> 4;
  const int l16  = lane & 15;
  const int wm   = (wave >> 2) * 64;          // 0 / 64
  const int wn   = (wave & 3) * 64;           // 0 / 64 / 128 / 192

  v8f acc[4][4];
#pragma unroll
  for (int i = 0; i < 4; ++i)
#pragma unroll
    for (int j = 0; j < 4; ++j) { v8f z = {}; acc[i][j] = z; }

  for (int kk = 0; kk < H_DIM; kk += 32) {
    if (kk + 32 < H_DIM) {                    // prefetch next K-slab
      __builtin_prefetch(wrow + kk + 32, 0, 3);
      if (arow_valid) __builtin_prefetch(xrow + kk + 32, 0, 3);
    }
    // stage A (gathered token rows), f32 -> bf16
    {
      u4v pa0, pa1;
#pragma unroll
      for (int j = 0; j < 2; ++j) {
        f4v a0 = {}, a1 = {};
        if (arow_valid) {
          a0 = *(const f4v*)(xrow + kk + j * 8);
          a1 = *(const f4v*)(xrow + kk + j * 8 + 4);
        }
        u4v p;
        p.x = pack_bf16(a0.x, a0.y); p.y = pack_bf16(a0.z, a0.w);
        p.z = pack_bf16(a1.x, a1.y); p.w = pack_bf16(a1.z, a1.w);
        if (j == 0) pa0 = p; else pa1 = p;
      }
      u4v* dst = (u4v*)&ldsA[arow * 20 + (kh >> 1)];
      dst[0] = pa0; dst[1] = pa1;             // 2x ds_store_b128
    }
    // stage B (weight rows, contiguous)
    {
#pragma unroll
      for (int j = 0; j < 4; ++j) {
        f4v b0 = *(const f4v*)(wrow + kk + j * 8);
        f4v b1 = *(const f4v*)(wrow + kk + j * 8 + 4);
        u4v p;
        p.x = pack_bf16(b0.x, b0.y); p.y = pack_bf16(b0.z, b0.w);
        p.z = pack_bf16(b1.x, b1.y); p.w = pack_bf16(b1.z, b1.w);
        ((u4v*)&ldsB[tid * 20])[j] = p;       // ds_store_b128
      }
    }
    __syncthreads();

    // B frags resident (4), A frag streamed (1 live) -> 16 WMMAs / K-step
    FragAB fb[4];
#pragma unroll
    for (int j = 0; j < 4; ++j) {
      const u4v* rp = (const u4v*)&ldsB[(wn + j * 16 + l16) * 20];
      fb[j].q[0] = rp[half];
      fb[j].q[1] = rp[2 + half];
    }
#pragma unroll
    for (int i = 0; i < 4; ++i) {
      FragAB fa;
      const u4v* rp = (const u4v*)&ldsA[(wm + i * 16 + l16) * 20];
      fa.q[0] = rp[half];                     // bytes 16*half
      fa.q[1] = rp[2 + half];                 // bytes 32 + 16*half
#pragma unroll
      for (int j = 0; j < 4; ++j)
        acc[i][j] = __builtin_amdgcn_wmma_f32_16x16x32_bf16(
            false, fa.v, false, fb[j].v, (short)0, acc[i][j], false, false);
    }
    __syncthreads();
  }

  // epilogue: C layout M = r + 8*half, N = lane%16; gate-scale, atomic add
#pragma unroll
  for (int i = 0; i < 4; ++i) {
#pragma unroll
    for (int r = 0; r < 8; ++r) {
      int row = wm + i * 16 + r + 8 * half;
      if (row < mrows) {
        int   tok = s_tok[row];
        float g   = s_gate[row];
        float* orow = out + (size_t)tok * H_DIM + n0 + wn + l16;
#pragma unroll
        for (int j = 0; j < 4; ++j)
          atomicAdd(orow + j * 16, g * acc[i][j][r]);
      }
    }
  }
}

// ---------------------------------------------------------------------------
extern "C" void kernel_launch(void* const* d_in, const int* in_sizes, int n_in,
                              void* d_out, int out_size, void* d_ws, size_t ws_size,
                              hipStream_t stream) {
  const float* tokens = (const float*)d_in[0];   // [T, H]
  const float* rw     = (const float*)d_in[1];   // [E, H]
  const float* rb     = (const float*)d_in[2];   // [E]
  const float* ew     = (const float*)d_in[3];   // [E, H, H]
  float* out = (float*)d_out;

  // workspace layout
  int*   counts   = (int*)d_ws;                  // [8]
  int*   offsets  = counts + 8;                  // [8]
  int*   cursor   = offsets + 8;                 // [8]
  int*   assign_e = cursor + 8;                  // [2T]
  float* assign_g = (float*)(assign_e + 2 * T_NUM);
  int*   btok     = (int*)(assign_g + 2 * T_NUM);
  float* bgate    = (float*)(btok + 2 * T_NUM);

  long long n4 = (long long)T_NUM * H_DIM / 4;
  moe_zero_kernel<<<(unsigned)((n4 + 255) / 256), 256, 0, stream>>>(out, counts, n4);
  moe_router_kernel<<<T_NUM / 8, 256, 0, stream>>>(tokens, rw, rb, assign_e, assign_g, counts);
  moe_scan_kernel<<<1, 32, 0, stream>>>(counts, offsets, cursor);
  moe_scatter_kernel<<<T_NUM / 256, 256, 0, stream>>>(assign_e, assign_g, cursor, btok, bgate);
  dim3 grid(H_DIM / 256, T_NUM / 128, E_NUM);    // (8, 64, 8); empty M-tiles exit early
  moe_gemm_kernel<<<grid, 256, 0, stream>>>(tokens, ew, counts, offsets, btok, bgate, out);
}
